// KernalAnsatz_65481071399682
// MI455X (gfx1250) — matrix operand order repaired and checked by reading, not compiled, not run on hardware
//
#include <hip/hip_runtime.h>

// ---------------------------------------------------------------------------
// RBF kernel matrix on MI455X (gfx1250):
//   out[i,j] = exp(-gamma * max(||x_i||^2 + ||y_j||^2 - 2 x_i.y_j, 0))
//
// Pass 1: convert X,Y (f32) -> bf16 copies in workspace (one-shot, RNE).
// Pass 2: row norms ||x||^2, ||y||^2 from f32 into workspace.
// Pass 3: tiled GEMM with V_WMMA_F32_16X16X32_BF16; 128x64 bf16 K-slabs
//         streamed global->LDS via GLOBAL_LOAD_ASYNC_TO_LDS_B128
//         (ASYNCcnt pipeline), double-buffered; fused exp epilogue.
// ---------------------------------------------------------------------------

typedef __attribute__((ext_vector_type(16))) __bf16       v16bf;
typedef __attribute__((ext_vector_type(8)))  float        v8f;
typedef __attribute__((ext_vector_type(4)))  float        v4f;
typedef __attribute__((ext_vector_type(4)))  unsigned int v4u;
typedef int v4i_n __attribute__((vector_size(16)));   // matches builtin sig

typedef __attribute__((address_space(1))) v4i_n* g_v4i_ptr;
typedef __attribute__((address_space(3))) v4i_n* l_v4i_ptr;

#define M_TOT 8192
#define N_TOT 8192
#define K_TOT 256
#define BM 128
#define BN 128
#define BK 64                    // K elements per LDS slab (2 WMMA K-steps)
#define KSTEPS (K_TOT / BK)      // 4
#define LDSW 36                  // uints per LDS row: 32 data (64 bf16) + 4 pad
#define GAMMA 1.0f

#if defined(__has_builtin)
#if __has_builtin(__builtin_amdgcn_global_load_async_to_lds_b128)
#define HAVE_ASYNC_LDS 1
#endif
#endif

#if defined(__has_builtin)
#if __has_builtin(__builtin_amdgcn_s_wait_asynccnt)
#define WAIT_ASYNC() __builtin_amdgcn_s_wait_asynccnt(0)
#else
#define WAIT_ASYNC() asm volatile("s_wait_asynccnt 0" ::: "memory")
#endif
#else
#define WAIT_ASYNC() asm volatile("s_wait_asynccnt 0" ::: "memory")
#endif

union FragU { v4u u[2]; v16bf h; };

__device__ __forceinline__ unsigned int pack2_bf16(float a, float b) {
    unsigned int ua = __float_as_uint(a);
    unsigned int ub = __float_as_uint(b);
    ua = (ua + 0x7FFFu + ((ua >> 16) & 1u)) >> 16;   // round-to-nearest-even
    ub = (ub + 0x7FFFu + ((ub >> 16) & 1u)) >> 16;
    return (ub << 16) | (ua & 0xFFFFu);
}

// ----------------------- pass 1: f32 -> bf16 convert -----------------------
__global__ void rbf_cvt_kernel(const float* __restrict__ X,
                               const float* __restrict__ Y,
                               unsigned int* __restrict__ Xb,   // bf16 pairs
                               unsigned int* __restrict__ Yb) {
    const size_t t = (size_t)blockIdx.x * blockDim.x + threadIdx.x;
    const size_t fi = t * 8;                 // 8 floats -> 4 packed uints
    const v4f* xa = (const v4f*)(X + fi);
    const v4f* ya = (const v4f*)(Y + fi);
    v4f a0 = xa[0], a1 = xa[1];
    v4f b0 = ya[0], b1 = ya[1];
    v4u pa = {pack2_bf16(a0.x, a0.y), pack2_bf16(a0.z, a0.w),
              pack2_bf16(a1.x, a1.y), pack2_bf16(a1.z, a1.w)};
    v4u pb = {pack2_bf16(b0.x, b0.y), pack2_bf16(b0.z, b0.w),
              pack2_bf16(b1.x, b1.y), pack2_bf16(b1.z, b1.w)};
    *(v4u*)(Xb + t * 4) = pa;
    *(v4u*)(Yb + t * 4) = pb;
}

// -------------------------- pass 2: row norms ------------------------------
__global__ void rbf_norms_kernel(const float* __restrict__ X,
                                 const float* __restrict__ Y,
                                 float* __restrict__ x2,
                                 float* __restrict__ y2) {
    int i = blockIdx.x * blockDim.x + threadIdx.x;   // 0..8191
    const v4f* xr = (const v4f*)(X + (size_t)i * K_TOT);
    const v4f* yr = (const v4f*)(Y + (size_t)i * K_TOT);
    float sx = 0.0f, sy = 0.0f;
#pragma unroll 8
    for (int k = 0; k < K_TOT / 4; ++k) {
        v4f a = xr[k];
        sx += a.x * a.x + a.y * a.y + a.z * a.z + a.w * a.w;
        v4f b = yr[k];
        sy += b.x * b.x + b.y * b.y + b.z * b.z + b.w * b.w;
    }
    x2[i] = sx;
    y2[i] = sy;
}

// ------------------------- pass 3: WMMA GEMM + exp -------------------------
__global__ __launch_bounds__(256)
void rbf_wmma_kernel(const unsigned short* __restrict__ Xb,
                     const unsigned short* __restrict__ Yb,
                     const float* __restrict__ x2,
                     const float* __restrict__ y2,
                     float* __restrict__ Out) {
    __shared__ unsigned int As[2][BM * LDSW];   // 18 KB per buffer
    __shared__ unsigned int Bs[2][BN * LDSW];

    const int tid  = threadIdx.x;
    const int lane = tid & 31;
    const int wave = tid >> 5;          // 0..7
    const int half = lane >> 4;         // lane half within wave32
    const int l16  = lane & 15;
    const int wm   = wave & 3;          // 4 waves along M -> 32-row strip
    const int wn   = wave >> 2;         // 2 waves along N -> 64-col strip

    const int blockRow = blockIdx.y * BM;
    const int blockCol = blockIdx.x * BN;

    // copy assignment: thread t owns 64 B (32 bf16) of each tile row:
    // row = t/2, K-segment = (t&1)*32 within the 64-wide slab.
    const int ldRow = tid >> 1;
    const int ldSeg = tid & 1;
    const unsigned short* xg = Xb + (size_t)(blockRow + ldRow) * K_TOT + ldSeg * 32;
    const unsigned short* yg = Yb + (size_t)(blockCol + ldRow) * K_TOT + ldSeg * 32;
    const int ldsOff = ldRow * LDSW + ldSeg * 16;   // uints

#if HAVE_ASYNC_LDS
    // GLOBAL_LOAD_ASYNC_TO_LDS_B128: INST_OFFSET is added to BOTH the global
    // and the LDS address, so one pointer pair + imm offsets covers the 64 B.
#define ASYNC_CP128(gp, lp, off)                                      \
    __builtin_amdgcn_global_load_async_to_lds_b128(                   \
        (g_v4i_ptr)(void*)(gp), (l_v4i_ptr)(void*)(lp), (off), 0)

#define TILE_COPY(buf, kt)                                            \
    do {                                                              \
        const unsigned short* ga = xg + (kt) * BK;                    \
        const unsigned short* gb = yg + (kt) * BK;                    \
        ASYNC_CP128(ga, &As[buf][ldsOff], 0);                         \
        ASYNC_CP128(ga, &As[buf][ldsOff], 16);                        \
        ASYNC_CP128(ga, &As[buf][ldsOff], 32);                        \
        ASYNC_CP128(ga, &As[buf][ldsOff], 48);                        \
        ASYNC_CP128(gb, &Bs[buf][ldsOff], 0);                         \
        ASYNC_CP128(gb, &Bs[buf][ldsOff], 16);                        \
        ASYNC_CP128(gb, &Bs[buf][ldsOff], 32);                        \
        ASYNC_CP128(gb, &Bs[buf][ldsOff], 48);                        \
    } while (0)
#define COPY_WAIT() WAIT_ASYNC()
#else
#define TILE_COPY(buf, kt)                                                    \
    do {                                                                      \
        const v4u* ga = (const v4u*)(xg + (kt) * BK);                         \
        const v4u* gb = (const v4u*)(yg + (kt) * BK);                         \
        v4u a0 = ga[0], a1 = ga[1], a2 = ga[2], a3 = ga[3];                   \
        v4u b0 = gb[0], b1 = gb[1], b2 = gb[2], b3 = gb[3];                   \
        *(__attribute__((address_space(3))) v4u*)&As[buf][ldsOff]      = a0;  \
        *(__attribute__((address_space(3))) v4u*)&As[buf][ldsOff + 4]  = a1;  \
        *(__attribute__((address_space(3))) v4u*)&As[buf][ldsOff + 8]  = a2;  \
        *(__attribute__((address_space(3))) v4u*)&As[buf][ldsOff + 12] = a3;  \
        *(__attribute__((address_space(3))) v4u*)&Bs[buf][ldsOff]      = b0;  \
        *(__attribute__((address_space(3))) v4u*)&Bs[buf][ldsOff + 4]  = b1;  \
        *(__attribute__((address_space(3))) v4u*)&Bs[buf][ldsOff + 8]  = b2;  \
        *(__attribute__((address_space(3))) v4u*)&Bs[buf][ldsOff + 12] = b3;  \
    } while (0)
#define COPY_WAIT() ((void)0)
#endif

    v8f acc[2][4];
#pragma unroll
    for (int mi = 0; mi < 2; ++mi)
#pragma unroll
        for (int ni = 0; ni < 4; ++ni)
            acc[mi][ni] = (v8f)0.0f;

    // one 32-K WMMA step over LDS slab `sub` (0/1) of buffers Ab/Bb
    auto compute32 = [&](const unsigned int* Ab, const unsigned int* Bb, int sub) {
        const int so = sub * 16;      // uint offset of 32-K sub-slab in row
        FragU fa[2];
#pragma unroll
        for (int mi = 0; mi < 2; ++mi) {
            // A fragment (16x32 bf16): half 0 holds K0-7 & K16-23,
            // half 1 holds K8-15 & K24-31 (ISA 7.12.2 layout).
            const int base = (wm * 32 + mi * 16 + l16) * LDSW + so;
            fa[mi].u[0] = *(const v4u*)&Ab[base + half * 4];
            fa[mi].u[1] = *(const v4u*)&Ab[base + 8 + half * 4];
        }
#pragma unroll
        for (int ni = 0; ni < 4; ++ni) {
            // B fragment (32x16 bf16): half 0 holds K0-7 & K16-23, half 1 K8-15 & K24-31.
            const int base = (wn * 64 + ni * 16 + l16) * LDSW + so;
            FragU fb;
            fb.u[0] = *(const v4u*)&Bb[base + half * 4];
            fb.u[1] = *(const v4u*)&Bb[base + 8 + half * 4];
#pragma unroll
            for (int mi = 0; mi < 2; ++mi) {
                acc[mi][ni] = __builtin_amdgcn_wmma_f32_16x16x32_bf16(
                    false, fa[mi].h, false, fb.h,
                    (short)0, acc[mi][ni], false, false);
            }
        }
    };

    // prologue: slab 0 -> LDS buffer 0
    TILE_COPY(0, 0);
    COPY_WAIT();
    __syncthreads();

#pragma unroll
    for (int kt = 0; kt < KSTEPS; ++kt) {
        const int cur = kt & 1;
        // prefetch next slab into the other buffer (free after last barrier)
        if (kt + 1 < KSTEPS) TILE_COPY(1 - cur, kt + 1);

        compute32(As[cur], Bs[cur], 0);
        compute32(As[cur], Bs[cur], 1);

        if (kt + 1 < KSTEPS) {
            COPY_WAIT();
            __syncthreads();
        }
    }

    // --- epilogue: sq = max(x2 + y2 - 2*s, 0); out = exp(-gamma*sq) ---
    float cy[4];
    int   col[4];
#pragma unroll
    for (int ni = 0; ni < 4; ++ni) {
        col[ni] = blockCol + wn * 64 + ni * 16 + l16;
        cy[ni]  = y2[col[ni]];
    }
#pragma unroll
    for (int mi = 0; mi < 2; ++mi) {
        // 8 row norms for this fragment row-group are contiguous: 2x b128
        const v4f* xr = (const v4f*)(x2 + blockRow + wm * 32 + mi * 16 + half * 8);
        v4f r0 = xr[0], r1 = xr[1];
        float rxv[8] = {r0.x, r0.y, r0.z, r0.w, r1.x, r1.y, r1.z, r1.w};
#pragma unroll
        for (int v = 0; v < 8; ++v) {
            const int row = blockRow + wm * 32 + mi * 16 + half * 8 + v;
            float* orow = Out + (size_t)row * N_TOT;
#pragma unroll
            for (int ni = 0; ni < 4; ++ni) {
                float s  = acc[mi][ni][v];
                float sq = fmaxf(rxv[v] + cy[ni] - 2.0f * s, 0.0f);
                orow[col[ni]] = __expf(-GAMMA * sq);
            }
        }
    }
}

// ------------------------------- launcher ----------------------------------
extern "C" void kernel_launch(void* const* d_in, const int* in_sizes, int n_in,
                              void* d_out, int out_size, void* d_ws, size_t ws_size,
                              hipStream_t stream) {
    (void)in_sizes; (void)n_in; (void)out_size; (void)ws_size;
    const float* x = (const float*)d_in[0];
    const float* y = (const float*)d_in[1];
    float* out = (float*)d_out;

    // workspace layout: Xb (4MB bf16) | Yb (4MB bf16) | x2 (32KB) | y2 (32KB)
    unsigned short* Xb = (unsigned short*)d_ws;
    unsigned short* Yb = Xb + (size_t)M_TOT * K_TOT;
    float* x2 = (float*)(Yb + (size_t)N_TOT * K_TOT);
    float* y2 = x2 + M_TOT;

    const int cvt_blocks = (int)(((size_t)M_TOT * K_TOT) / (256 * 8));  // 1024
    rbf_cvt_kernel<<<cvt_blocks, 256, 0, stream>>>(x, y,
                                                   (unsigned int*)Xb,
                                                   (unsigned int*)Yb);
    rbf_norms_kernel<<<M_TOT / 256, 256, 0, stream>>>(x, y, x2, y2);

    dim3 grid(N_TOT / BN, M_TOT / BM);   // 64 x 64 blocks
    rbf_wmma_kernel<<<grid, 256, 0, stream>>>(Xb, Yb, x2, y2, out);
}
